// LSTMLayer_42941083025995
// MI455X (gfx1250) — matrix operand order repaired and checked by reading, not compiled, not run on hardware
//
#include <hip/hip_runtime.h>

// LSTM layer for MI455X (gfx1250, wave32, WMMA).
// gates = [x_t | h_{t-1}] @ [W_ih ; W_hh]^T + (b_ih+b_hh), fused per step.
// bf16 WMMA 16x16x32, f32 accumulate. Weight image (12.6 MB) stays in L2.
// K-loop is register double-buffered so next iteration's b128 loads overlap
// the current iteration's 4 WMMAs. 64 blocks x 4 waves = 256 waves/step.

#define T_STEPS 1024
#define BATCH   64
#define IN_DIM  512
#define HID     1024
#define GATES   (4 * HID)          // 4096
#define KDIM    (IN_DIM + HID)     // 1536

typedef __attribute__((ext_vector_type(16))) __bf16 v16bf;
typedef __attribute__((ext_vector_type(8)))  float  v8f;

__device__ __forceinline__ unsigned short f2bfb(float f) {
    unsigned u = __float_as_uint(f);
    u += 0x7FFFu + ((u >> 16) & 1u);          // round-to-nearest-even
    return (unsigned short)(u >> 16);
}
__device__ __forceinline__ float sigm(float x)      { return 1.0f / (1.0f + __expf(-x)); }
__device__ __forceinline__ float tanh_fast(float x) { return 1.0f - 2.0f / (__expf(2.0f * x) + 1.0f); }

// Build fused bf16 weight image Wbf[4H][KDIM] = [w_ih | w_hh], and bias = b_ih + b_hh.
__global__ void build_weights_kernel(const float* __restrict__ w_ih,
                                     const float* __restrict__ w_hh,
                                     const float* __restrict__ b_ih,
                                     const float* __restrict__ b_hh,
                                     unsigned short* __restrict__ Wbf,
                                     float* __restrict__ bias) {
    int idx = blockIdx.x * blockDim.x + threadIdx.x;
    if (idx < GATES * KDIM) {
        int n = idx / KDIM, k = idx % KDIM;
        float v = (k < IN_DIM) ? w_ih[n * IN_DIM + k] : w_hh[n * HID + (k - IN_DIM)];
        Wbf[idx] = f2bfb(v);
    }
    if (idx < GATES) bias[idx] = b_ih[idx] + b_hh[idx];
}

// Bulk f32 -> bf16 conversion of x (4 elements per thread, vectorized loads).
__global__ void convert_x_kernel(const float* __restrict__ x,
                                 unsigned short* __restrict__ xbf) {
    size_t i4 = (size_t)blockIdx.x * blockDim.x + threadIdx.x;   // quad index
    if (i4 < (size_t)T_STEPS * BATCH * IN_DIM / 4) {
        float4 f = *(const float4*)(x + i4 * 4);
        ushort4 o;
        o.x = f2bfb(f.x); o.y = f2bfb(f.y); o.z = f2bfb(f.z); o.w = f2bfb(f.w);
        *(ushort4*)(xbf + i4 * 4) = o;
    }
}

__global__ void init_state_kernel(const float* __restrict__ h0,
                                  const float* __restrict__ c0,
                                  unsigned short* __restrict__ hbuf0,
                                  float* __restrict__ c_state) {
    int idx = blockIdx.x * blockDim.x + threadIdx.x;
    if (idx < BATCH * HID) {
        hbuf0[idx]   = f2bfb(h0[idx]);
        c_state[idx] = c0[idx];
    }
}

// One recurrence step. 64 blocks x 128 threads = 256 waves (1 wave/SIMD shape).
// Wave (mtile, jtile): batch rows [16*mtile, +16), hidden cols [16*jtile, +16),
// 4 accumulators (i,f,g,o gates) over K = 1536, register double-buffered.
template <bool XPRE>
__global__ __launch_bounds__(128) void lstm_step_kernel(
    const void*           __restrict__ x_t,    // [B][I]  bf16 (XPRE) or f32
    const unsigned short* __restrict__ hprev,  // [B][H]  bf16
    unsigned short*       __restrict__ hnext,  // [B][H]  bf16
    const unsigned short* __restrict__ Wbf,    // [4H][KDIM] bf16
    const float*          __restrict__ bias,   // [4H]
    float*                __restrict__ c_state,// [B][H]
    float*                __restrict__ out_t)  // [B][H]
{
    const int lane  = threadIdx.x & 31;
    const int wave  = threadIdx.x >> 5;
    const int tile  = blockIdx.x * 4 + wave;   // 0..255, exact cover, no divergence
    const int mtile = tile & 3;
    const int jtile = tile >> 2;

    const int m16 = lane & 15;                 // A-matrix row within tile (both lane halves)
    const int hs  = lane >> 4;                 // lane-half selects K sub-block
    const int b   = mtile * 16 + m16;          // batch row

    v8f zero = {0.f, 0.f, 0.f, 0.f, 0.f, 0.f, 0.f, 0.f};
    v8f acc[4] = {zero, zero, zero, zero};

    // B-fragment base pointers: column n = lane&15 -> weight row g*H + jtile*16 + n;
    // lane-half picks K offset +16; 16 contiguous bf16 per lane (32B aligned).
    const unsigned short* wptr[4];
#pragma unroll
    for (int g = 0; g < 4; ++g) {
        int row = g * HID + jtile * 16 + (lane & 15);
        wptr[g] = Wbf + (size_t)row * KDIM + hs * 16;
    }

    // A-source base pointers, indexable directly by k0.
    const unsigned short* xb16 = XPRE ? ((const unsigned short*)x_t + (size_t)b * IN_DIM) : nullptr;
    const float*          xb32 = XPRE ? nullptr : ((const float*)x_t + (size_t)b * IN_DIM);
    const unsigned short* hb   = hprev + (size_t)b * HID - IN_DIM;   // valid for k0 >= IN_DIM

    union AF { v16bf v; unsigned u[8]; };

    // A fragment, contiguous K layout (matches B): pair p holds K = kb(p), kb(p)+1 with
    // kb(p) = 2p + 16*lanehalf.
    auto loadA = [&](int k0, AF& r) {
        if (XPRE || k0 >= IN_DIM) {
            const unsigned short* asrc = (k0 < IN_DIM) ? (xb16 + k0) : (hb + k0);
#pragma unroll
            for (int p = 0; p < 8; ++p) {
                int kb = 2 * p + hs * 16;
                r.u[p] = *(const unsigned*)(asrc + kb);   // merges into 2x b128
            }
        } else {
            const float* ax = xb32 + k0;                  // f32 fallback path
#pragma unroll
            for (int p = 0; p < 8; ++p) {
                int kb = 2 * p + hs * 16;
                float2 f = *(const float2*)(ax + kb);
                r.u[p] = (unsigned)f2bfb(f.x) | ((unsigned)f2bfb(f.y) << 16);
            }
        }
    };
    auto loadB = [&](int k0, v16bf* dst) {
#pragma unroll
        for (int g = 0; g < 4; ++g) dst[g] = *(const v16bf*)(wptr[g] + k0);
    };

    // Register double-buffered K loop: KDIM = 1536 = 24 * 64.
    AF    a[2];
    v16bf bb[2][4];
    loadA(0, a[0]);
    loadB(0, bb[0]);
    for (int k0 = 0; k0 < KDIM; k0 += 64) {
        loadA(k0 + 32, a[1]);                 // prefetch odd phase (always in range)
        loadB(k0 + 32, bb[1]);
#pragma unroll
        for (int g = 0; g < 4; ++g)
            acc[g] = __builtin_amdgcn_wmma_f32_16x16x32_bf16(
                false, a[0].v, false, bb[0][g], (short)0, acc[g], false, false);
        if (k0 + 64 < KDIM) {                 // prefetch next even phase
            loadA(k0 + 64, a[0]);
            loadB(k0 + 64, bb[0]);
        }
#pragma unroll
        for (int g = 0; g < 4; ++g)
            acc[g] = __builtin_amdgcn_wmma_f32_16x16x32_bf16(
                false, a[1].v, false, bb[1][g], (short)0, acc[g], false, false);
    }

    // Lane-local LSTM cell update. C/D layout: lane -> column n, VGPR r -> row r+8*lanehalf.
    const int n = lane & 15;
    const int j = jtile * 16 + n;
    const float bi = bias[j];
    const float bf = bias[HID + j];
    const float bg = bias[2 * HID + j];
    const float bo = bias[3 * HID + j];
#pragma unroll
    for (int r = 0; r < 8; ++r) {
        int brow = mtile * 16 + r + 8 * hs;
        size_t o = (size_t)brow * HID + j;
        float ig = sigm(acc[0][r] + bi);
        float fg = sigm(acc[1][r] + bf);
        float gg = tanh_fast(acc[2][r] + bg);
        float og = sigm(acc[3][r] + bo);
        float c  = fg * c_state[o] + ig * gg;
        float h  = og * tanh_fast(c);
        c_state[o] = c;
        out_t[o]   = h;
        hnext[o]   = f2bfb(h);
    }
}

__global__ void finalize_kernel(const float* __restrict__ c_state, float* __restrict__ out) {
    int idx = blockIdx.x * blockDim.x + threadIdx.x;
    if (idx < BATCH * HID) {
        size_t base = (size_t)T_STEPS * BATCH * HID;
        out[base + idx] = out[(size_t)(T_STEPS - 1) * BATCH * HID + idx];   // h_f
        out[base + BATCH * HID + idx] = c_state[idx];                       // c_f
    }
}

extern "C" void kernel_launch(void* const* d_in, const int* in_sizes, int n_in,
                              void* d_out, int out_size, void* d_ws, size_t ws_size,
                              hipStream_t stream) {
    const float* x    = (const float*)d_in[0];
    const float* h0   = (const float*)d_in[1];
    const float* c0   = (const float*)d_in[2];
    const float* w_ih = (const float*)d_in[3];
    const float* w_hh = (const float*)d_in[4];
    const float* b_ih = (const float*)d_in[5];
    const float* b_hh = (const float*)d_in[6];
    float* out = (float*)d_out;

    // Workspace layout, re-initialized every call (deterministic).
    char* ws = (char*)d_ws;
    size_t off = 0;
    auto alloc = [&](size_t bytes) -> char* {
        char* p = ws + off;
        off = (off + bytes + 255) & ~(size_t)255;
        return p;
    };
    unsigned short* Wbf  = (unsigned short*)alloc((size_t)GATES * KDIM * 2);   // 12.6 MB
    float*          bias = (float*)alloc((size_t)GATES * 4);
    unsigned short* hbuf = (unsigned short*)alloc((size_t)2 * BATCH * HID * 2);
    float*          cst  = (float*)alloc((size_t)BATCH * HID * 4);
    unsigned short* xbf  = (unsigned short*)alloc((size_t)T_STEPS * BATCH * IN_DIM * 2); // 64 MB
    const bool xpre = (ws_size >= off);        // enough room for pre-converted x?

    build_weights_kernel<<<(GATES * KDIM + 255) / 256, 256, 0, stream>>>(
        w_ih, w_hh, b_ih, b_hh, Wbf, bias);
    init_state_kernel<<<(BATCH * HID + 255) / 256, 256, 0, stream>>>(h0, c0, hbuf, cst);

    if (xpre) {
        size_t nquads = (size_t)T_STEPS * BATCH * IN_DIM / 4;
        convert_x_kernel<<<(unsigned)((nquads + 255) / 256), 256, 0, stream>>>(x, xbf);
        for (int t = 0; t < T_STEPS; ++t) {
            lstm_step_kernel<true><<<64, 128, 0, stream>>>(
                xbf + (size_t)t * BATCH * IN_DIM,
                hbuf + (size_t)(t & 1) * BATCH * HID,
                hbuf + (size_t)((t + 1) & 1) * BATCH * HID,
                Wbf, bias, cst,
                out + (size_t)t * BATCH * HID);
        }
    } else {
        for (int t = 0; t < T_STEPS; ++t) {
            lstm_step_kernel<false><<<64, 128, 0, stream>>>(
                x + (size_t)t * BATCH * IN_DIM,
                hbuf + (size_t)(t & 1) * BATCH * HID,
                hbuf + (size_t)((t + 1) & 1) * BATCH * HID,
                Wbf, bias, cst,
                out + (size_t)t * BATCH * HID);
        }
    }

    finalize_kernel<<<(BATCH * HID + 255) / 256, 256, 0, stream>>>(cst, out);
}
